// GMAN_59158879535138
// MI455X (gfx1250) — matrix operand order repaired and checked by reading, not compiled, not run on hardware
//
#include <hip/hip_runtime.h>
#include <hip/hip_bf16.h>
#include <math.h>

typedef __attribute__((ext_vector_type(16))) _Float16 v16h;
typedef __attribute__((ext_vector_type(8)))  _Float16 v8h;
typedef __attribute__((ext_vector_type(8)))  float    v8f;

#define NNODES 325
#define PSTEPS 12
#define BATCH  8
#define HEADS  8
#define HDIM   8
#define DMODEL 64
#define MROWS  (BATCH * PSTEPS * NNODES)   // 31200 (divisible by 16)
#define RSQRT_D 0.35355339059327f
#define KMAX   128
#define KPAD   8                           // LDS pad (halves) to stagger banks

// ---------------------------------------------------------------------------
// WMMA GEMM, N fixed at 64:  Y[M,64] = act(A[M,K] @ W[K,64] + bias[64])
// Requirements: K % 32 == 0, K <= 128. Each wave computes a 16x64 strip:
// one A fragment feeds FOUR v_wmma ops. Full W staged in LDS as f16
// (<= 17 KB of the 320 KB WGP LDS), layout [col][k] (+8-half pad).
// Compile-time variants: HAS_BIAS (no dynamic null check), RELU,
// FULL (M % 64 == 0: no row clamp, no store guards).
// ---------------------------------------------------------------------------
template<bool HAS_BIAS, bool RELU, bool FULL>
__global__ __launch_bounds__(128)
void gemm_wmma_kernel(const float* __restrict__ A, const float* __restrict__ W,
                      const float* __restrict__ bias, float* __restrict__ Y,
                      int M, int K)
{
    const int lane    = threadIdx.x & 31;
    const int wave    = threadIdx.x >> 5;
    const int row0    = (blockIdx.x * 4 + wave) * 16;
    const int halfSel = lane >> 4;          // 0: lanes 0-15, 1: lanes 16-31
    const int lrow    = lane & 15;
    const int kstride = K + KPAD;

    // Warm L2 for this wave's A rows while the block stages W.
    {
        int prow = FULL ? (row0 + lrow) : min(row0 + lrow, M - 1);
        __builtin_prefetch(&A[(size_t)prow * K], 0, 0);
    }

    // ---- stage full W[K,64] into LDS as f16, layout [col][k] (padded) ----
    __shared__ _Float16 Bs[DMODEL * (KMAX + KPAD)];
    for (int idx = threadIdx.x; idx < K * DMODEL; idx += 128) {
        int c = idx & 63;                   // coalesced read of W rows
        int k = idx >> 6;
        Bs[c * kstride + k] = (_Float16)W[(size_t)k * DMODEL + c];
    }
    __syncthreads();
    if (!FULL && row0 >= M) return;         // after barrier: safe

    // A fragment: row = lane&15, two 32B runs per 32-chunk
    const int arow = FULL ? (row0 + lrow) : min(row0 + lrow, M - 1);
    const int akb  = halfSel * 8;           // A k-base (ISA 16-bit 16x32 layout)
    const int bkb  = halfSel * 8;           // B k-base (same layout as A)
    const float4* A4 = (const float4*)A;
    const size_t rowbase4 = (size_t)arow * (K >> 2);

    v8f acc[4] = {};
    for (int kk = 0; kk < K; kk += 32) {
        const int b0 = (kk + akb) >> 2;
        float4 a0 = A4[rowbase4 + b0 + 0];   // k = akb+0..3
        float4 a1 = A4[rowbase4 + b0 + 1];   // k = akb+4..7
        float4 a2 = A4[rowbase4 + b0 + 4];   // k = akb+16..19
        float4 a3 = A4[rowbase4 + b0 + 5];   // k = akb+20..23
        v16h a;
        a[0]  = (_Float16)a0.x; a[1]  = (_Float16)a0.y;
        a[2]  = (_Float16)a0.z; a[3]  = (_Float16)a0.w;
        a[4]  = (_Float16)a1.x; a[5]  = (_Float16)a1.y;
        a[6]  = (_Float16)a1.z; a[7]  = (_Float16)a1.w;
        a[8]  = (_Float16)a2.x; a[9]  = (_Float16)a2.y;
        a[10] = (_Float16)a2.z; a[11] = (_Float16)a2.w;
        a[12] = (_Float16)a3.x; a[13] = (_Float16)a3.y;
        a[14] = (_Float16)a3.z; a[15] = (_Float16)a3.w;

#pragma unroll
        for (int t = 0; t < 4; ++t) {        // four 16-col tiles of the strip
            const _Float16* bp = &Bs[(t * 16 + lrow) * kstride + kk + bkb];
            v8h bl = *(const v8h*)(bp);      // 16B-aligned ds_load_b128
            v8h bh = *(const v8h*)(bp + 16);
            v16h b;
#pragma unroll
            for (int h = 0; h < 8; ++h) { b[h] = bl[h]; b[h + 8] = bh[h]; }
            acc[t] = __builtin_amdgcn_wmma_f32_16x16x32_f16(false, a, false, b,
                                                            (short)0, acc[t],
                                                            false, false);
        }
    }

    // Epilogue: unguarded, coalesced bias loads (one per column tile).
    float bv[4] = {0.0f, 0.0f, 0.0f, 0.0f};
    if (HAS_BIAS) {
#pragma unroll
        for (int t = 0; t < 4; ++t) bv[t] = bias[t * 16 + lrow];
    }
#pragma unroll
    for (int t = 0; t < 4; ++t) {
#pragma unroll
        for (int v = 0; v < 8; ++v) {
            int m = row0 + v + halfSel * 8;  // C/D: M = vgpr (+8 hi lanes)
            if (!FULL && m >= M) continue;
            int n = t * 16 + lrow;           // N = tile*16 + lane&15
            float val = acc[t][v] + bv[t];
            if (RELU) val = fmaxf(val, 0.0f);
            Y[(size_t)m * DMODEL + n] = val;
        }
    }
}

// ---------------------------------------------------------------------------
// Elementwise kernels
// ---------------------------------------------------------------------------
__global__ void fcin_hidden_kernel(const float* __restrict__ X,
                                   const float* __restrict__ W1,
                                   const float* __restrict__ b1,
                                   float* __restrict__ H, int M)
{
    int i = blockIdx.x * blockDim.x + threadIdx.x;
    if (i >= M * DMODEL) return;
    int row = i >> 6, j = i & 63;
    H[i] = fmaxf(X[row] * W1[j] + b1[j], 0.0f);
}

// TE one-hot @ W1  ==  W1[day] + W1[7+tod]  (row gather), then ReLU(+b1)
__global__ void te_hidden_kernel(const int* __restrict__ TE,
                                 const float* __restrict__ W1,
                                 const float* __restrict__ b1,
                                 float* __restrict__ H, int R)
{
    int i = blockIdx.x * blockDim.x + threadIdx.x;
    if (i >= R * DMODEL) return;
    int r = i >> 6, j = i & 63;
    int day = TE[r * 2 + 0];
    int tod = TE[r * 2 + 1];
    float v = W1[(size_t)day * DMODEL + j] + W1[(size_t)(7 + tod) * DMODEL + j] + b1[j];
    H[i] = fmaxf(v, 0.0f);
}

// STE = se_out[n] + te_out[b,t]; split directly into flattened STE_P / STE_Q
__global__ void ste_combine_kernel(const float* __restrict__ se,
                                   const float* __restrict__ te,
                                   float* __restrict__ steP,
                                   float* __restrict__ steQ)
{
    int i = blockIdx.x * blockDim.x + threadIdx.x;
    int total = BATCH * 24 * NNODES * DMODEL;
    if (i >= total) return;
    int j = i & 63;
    int t = i >> 6;
    int n = t % NNODES; t /= NNODES;
    int step = t % 24;
    int b    = t / 24;
    float v = se[(size_t)n * DMODEL + j] + te[(size_t)(b * 24 + step) * DMODEL + j];
    if (step < PSTEPS)
        steP[(((size_t)(b * PSTEPS + step)) * NNODES + n) * DMODEL + j] = v;
    else
        steQ[(((size_t)(b * PSTEPS + (step - 12))) * NNODES + n) * DMODEL + j] = v;
}

__global__ void concat_kernel(const float* __restrict__ X,
                              const float* __restrict__ STE,
                              float* __restrict__ XC, int M)
{
    int i = blockIdx.x * blockDim.x + threadIdx.x;
    if (i >= M * DMODEL) return;
    int row = i >> 6, j = i & 63;
    XC[(size_t)row * 128 + j]      = X[i];
    XC[(size_t)row * 128 + 64 + j] = STE[i];
}

__global__ void gate_kernel(const float* __restrict__ gs, const float* __restrict__ gt,
                            const float* __restrict__ HS, const float* __restrict__ HT,
                            float* __restrict__ Hout, int total)
{
    int i = blockIdx.x * blockDim.x + threadIdx.x;
    if (i >= total) return;
    float z = 1.0f / (1.0f + __expf(-(gs[i] + gt[i])));
    Hout[i] = z * HS[i] + (1.0f - z) * HT[i];
}

__global__ void add_kernel(float* __restrict__ X, const float* __restrict__ Yv, int total)
{
    int i = blockIdx.x * blockDim.x + threadIdx.x;
    if (i >= total) return;
    X[i] += Yv[i];
}

__global__ void out_dot_kernel(const float* __restrict__ H,
                               const float* __restrict__ W2,
                               const float* __restrict__ b2,
                               float* __restrict__ out, int M)
{
    int row = blockIdx.x * blockDim.x + threadIdx.x;
    if (row >= M) return;
    float s = b2[0];
#pragma unroll
    for (int j = 0; j < DMODEL; ++j)
        s += H[(size_t)row * DMODEL + j] * W2[j];
    out[row] = s;
}

// ---------------------------------------------------------------------------
// Spatial attention: per (b,p,head), 325 queries over 325 keys, d=8.
// K/V staged in LDS (float4 loads); one thread per query; online softmax.
// ---------------------------------------------------------------------------
__global__ __launch_bounds__(352)
void spatial_attn_kernel(const float* __restrict__ Q, const float* __restrict__ Kb,
                         const float* __restrict__ V, float* __restrict__ O)
{
    const int bp   = blockIdx.x / HEADS;   // b*P + p
    const int head = blockIdx.x % HEADS;
    __shared__ __align__(16) float Ks[NNODES * HDIM];
    __shared__ __align__(16) float Vs[NNODES * HDIM];
    // 2 x float4 per key row (head slice is 32B-aligned in a 64-float row)
    for (int idx = threadIdx.x; idx < NNODES * 2; idx += blockDim.x) {
        int m = idx >> 1, part = idx & 1;
        size_t base = ((size_t)bp * NNODES + m) * DMODEL + head * HDIM;
        ((float4*)Ks)[idx] = ((const float4*)(Kb + base))[part];
        ((float4*)Vs)[idx] = ((const float4*)(V + base))[part];
    }
    __syncthreads();
    const int n = threadIdx.x;
    if (n >= NNODES) return;

    float q[HDIM];
    size_t qbase = ((size_t)bp * NNODES + n) * DMODEL + head * HDIM;
    {
        float4 q0 = ((const float4*)(Q + qbase))[0];
        float4 q1 = ((const float4*)(Q + qbase))[1];
        q[0] = q0.x * RSQRT_D; q[1] = q0.y * RSQRT_D;
        q[2] = q0.z * RSQRT_D; q[3] = q0.w * RSQRT_D;
        q[4] = q1.x * RSQRT_D; q[5] = q1.y * RSQRT_D;
        q[6] = q1.z * RSQRT_D; q[7] = q1.w * RSQRT_D;
    }

    float mmax = -3.0e38f, ssum = 0.0f, acc[HDIM];
#pragma unroll
    for (int d = 0; d < HDIM; ++d) acc[d] = 0.0f;

    for (int m = 0; m < NNODES; ++m) {
        float s = 0.0f;
#pragma unroll
        for (int d = 0; d < HDIM; ++d) s += q[d] * Ks[m * HDIM + d];
        float nm    = fmaxf(mmax, s);
        float scale = __expf(mmax - nm);
        float w     = __expf(s - nm);
        ssum = ssum * scale + w;
#pragma unroll
        for (int d = 0; d < HDIM; ++d) acc[d] = acc[d] * scale + w * Vs[m * HDIM + d];
        mmax = nm;
    }
    float inv = 1.0f / ssum;
#pragma unroll
    for (int d = 0; d < HDIM; ++d) O[qbase + d] = acc[d] * inv;
}

// ---------------------------------------------------------------------------
// Small attention (temporal causal / transform): per (b,head,n,query),
// 12 keys, d=8. One thread per query.
// ---------------------------------------------------------------------------
__global__ void small_attn_kernel(const float* __restrict__ Q, const float* __restrict__ Kb,
                                  const float* __restrict__ V, float* __restrict__ O,
                                  int causal)
{
    int i = blockIdx.x * blockDim.x + threadIdx.x;
    int total = BATCH * HEADS * NNODES * PSTEPS;
    if (i >= total) return;
    int p = i % PSTEPS; int t = i / PSTEPS;
    int n = t % NNODES; t /= NNODES;
    int head = t % HEADS;
    int b    = t / HEADS;

    float q[HDIM];
    size_t qbase = (((size_t)(b * PSTEPS + p)) * NNODES + n) * DMODEL + head * HDIM;
#pragma unroll
    for (int d = 0; d < HDIM; ++d) q[d] = Q[qbase + d] * RSQRT_D;

    int nkeys = causal ? (p + 1) : PSTEPS;
    float s[PSTEPS];
    float mx = -3.0e38f;
    for (int m = 0; m < nkeys; ++m) {
        size_t kbase = (((size_t)(b * PSTEPS + m)) * NNODES + n) * DMODEL + head * HDIM;
        float sv = 0.0f;
#pragma unroll
        for (int d = 0; d < HDIM; ++d) sv += q[d] * Kb[kbase + d];
        s[m] = sv;
        mx = fmaxf(mx, sv);
    }
    float ssum = 0.0f;
    for (int m = 0; m < nkeys; ++m) { s[m] = __expf(s[m] - mx); ssum += s[m]; }
    float inv = 1.0f / ssum;
    float acc[HDIM];
#pragma unroll
    for (int d = 0; d < HDIM; ++d) acc[d] = 0.0f;
    for (int m = 0; m < nkeys; ++m) {
        size_t vbase = (((size_t)(b * PSTEPS + m)) * NNODES + n) * DMODEL + head * HDIM;
        float w = s[m] * inv;
#pragma unroll
        for (int d = 0; d < HDIM; ++d) acc[d] += w * V[vbase + d];
    }
#pragma unroll
    for (int d = 0; d < HDIM; ++d) O[qbase + d] = acc[d];
}

// ---------------------------------------------------------------------------
// Host orchestration
// ---------------------------------------------------------------------------
static inline void gemm(hipStream_t st, const float* A, const float* W, const float* bias,
                        float* Y, int M, int K, int act)
{
    dim3 grid((M + 63) / 64);
    const bool full = (M % 64) == 0;
    if (bias) {
        if (act) {
            if (full) gemm_wmma_kernel<true,  true,  true ><<<grid, 128, 0, st>>>(A, W, bias, Y, M, K);
            else      gemm_wmma_kernel<true,  true,  false><<<grid, 128, 0, st>>>(A, W, bias, Y, M, K);
        } else {
            if (full) gemm_wmma_kernel<true,  false, true ><<<grid, 128, 0, st>>>(A, W, bias, Y, M, K);
            else      gemm_wmma_kernel<true,  false, false><<<grid, 128, 0, st>>>(A, W, bias, Y, M, K);
        }
    } else {
        if (act) {
            if (full) gemm_wmma_kernel<false, true,  true ><<<grid, 128, 0, st>>>(A, W, bias, Y, M, K);
            else      gemm_wmma_kernel<false, true,  false><<<grid, 128, 0, st>>>(A, W, bias, Y, M, K);
        } else {
            if (full) gemm_wmma_kernel<false, false, true ><<<grid, 128, 0, st>>>(A, W, bias, Y, M, K);
            else      gemm_wmma_kernel<false, false, false><<<grid, 128, 0, st>>>(A, W, bias, Y, M, K);
        }
    }
}

static inline void ew(int total, int& blocks) { blocks = (total + 255) / 256; }

extern "C" void kernel_launch(void* const* d_in, const int* in_sizes, int n_in,
                              void* d_out, int out_size, void* d_ws, size_t ws_size,
                              hipStream_t stream)
{
    (void)in_sizes; (void)n_in; (void)out_size; (void)ws_size;
    auto F = [&](int i) { return (const float*)d_in[i]; };
    const int* TE = (const int*)d_in[2];

    // Parameter indices (insertion-order flattening of setup_inputs()):
    // 0 X, 1 SE, 2 TE
    // fc_in: 3 W1, 4 b1, 5 W2, 6 b2
    // ste.se: 7..10 ; ste.te: 11..14
    // enc block: base 15 ; trans: base 42 ; dec block: base 52
    // fc_out: 79 W1, 80 b1, 81 W2, 82 b2
    const int ENC = 15, TRANS = 42, DEC = 52, FCO = 79;
    // block offsets: sa q/k/v/fc = +0..9, ta = +10..19,
    // gf: xsW=+20, xtW=+21, xtb=+22, hW1=+23, hb1=+24, hW2=+25, hb2=+26

    const int M = MROWS;
    const size_t A = (size_t)M * DMODEL;     // 1,996,800 floats
    float* ws   = (float*)d_ws;
    float* xcur = ws;                        // current activation
    float* steP = ws + 1 * A;
    float* steQ = ws + 2 * A;
    float* xcat = ws + 3 * A;                // (M,128) -> 2*A
    float* qb   = ws + 5 * A;
    float* kb   = ws + 6 * A;
    float* vb   = ws + 7 * A;
    float* att  = ws + 8 * A;
    float* hb   = ws + 9 * A;
    float* hs   = ws + 10 * A;
    float* ht   = ws + 11 * A;
    float* smem = ws + 12 * A;               // small area
    float* se_h   = smem;                    // 325*64
    float* se_out = smem + 20800;            // 325*64
    float* te_h   = smem + 41600;            // 192*64
    float* te_out = smem + 53888;            // 192*64

    int blk;

    // ---- fc_in: X[...,None] -> (M,64) ----
    ew(M * DMODEL, blk);
    fcin_hidden_kernel<<<blk, 256, 0, stream>>>(F(0), F(3), F(4), hb, M);
    gemm(stream, hb, F(5), F(6), xcur, M, DMODEL, 0);

    // ---- STE ----
    gemm(stream, F(1), F(7), F(8), se_h, NNODES, DMODEL, 1);
    gemm(stream, se_h, F(9), F(10), se_out, NNODES, DMODEL, 0);
    const int R = BATCH * 24;
    ew(R * DMODEL, blk);
    te_hidden_kernel<<<blk, 256, 0, stream>>>(TE, F(11), F(12), te_h, R);
    gemm(stream, te_h, F(13), F(14), te_out, R, DMODEL, 0);
    ew(BATCH * 24 * NNODES * DMODEL, blk);
    ste_combine_kernel<<<blk, 256, 0, stream>>>(se_out, te_out, steP, steQ);

    // ---- encoder/decoder block ----
    auto run_att = [&](int base, bool spatial, const float* src, int cin) {
        gemm(stream, src, F(base + 0), F(base + 1), qb, M, cin, 1);
        gemm(stream, src, F(base + 2), F(base + 3), kb, M, cin, 1);
        gemm(stream, src, F(base + 4), F(base + 5), vb, M, cin, 1);
        if (spatial) {
            spatial_attn_kernel<<<BATCH * PSTEPS * HEADS, 352, 0, stream>>>(qb, kb, vb, att);
        } else {
            int total = BATCH * HEADS * NNODES * PSTEPS;
            small_attn_kernel<<<(total + 255) / 256, 256, 0, stream>>>(qb, kb, vb, att, 1);
        }
    };
    auto run_fc2 = [&](int wbase, const float* src, float* dst) {
        gemm(stream, src, F(wbase + 0), F(wbase + 1), hb, M, DMODEL, 1);
        gemm(stream, hb, F(wbase + 2), F(wbase + 3), dst, M, DMODEL, 0);
    };
    auto run_block = [&](int base, const float* steFlat) {
        int blk2;
        ew(M * DMODEL, blk2);
        concat_kernel<<<blk2, 256, 0, stream>>>(xcur, steFlat, xcat, M);
        // spatial attention -> hs
        run_att(base + 0, true, xcat, 128);
        run_fc2(base + 6, att, hs);
        // temporal attention -> ht
        run_att(base + 10, false, xcat, 128);
        run_fc2(base + 16, att, ht);
        // gated fusion: reuse qb/kb as gate pre-activations, att as gated H
        gemm(stream, hs, F(base + 20), nullptr, qb, M, DMODEL, 0);
        gemm(stream, ht, F(base + 21), F(base + 22), kb, M, DMODEL, 0);
        gate_kernel<<<blk2, 256, 0, stream>>>(qb, kb, hs, ht, att, M * DMODEL);
        run_fc2(base + 23, att, vb);                 // vb = fc2(gated)
        add_kernel<<<blk2, 256, 0, stream>>>(xcur, vb, M * DMODEL);  // residual
    };

    run_block(ENC, steP);

    // ---- transform attention (Q from STE_Q, K from STE_P, V from X) ----
    gemm(stream, steQ, F(TRANS + 0), F(TRANS + 1), qb, M, DMODEL, 1);
    gemm(stream, steP, F(TRANS + 2), F(TRANS + 3), kb, M, DMODEL, 1);
    gemm(stream, xcur, F(TRANS + 4), F(TRANS + 5), vb, M, DMODEL, 1);
    {
        int total = BATCH * HEADS * NNODES * PSTEPS;
        small_attn_kernel<<<(total + 255) / 256, 256, 0, stream>>>(qb, kb, vb, att, 0);
    }
    gemm(stream, att, F(TRANS + 6), F(TRANS + 7), hb, M, DMODEL, 1);
    gemm(stream, hb, F(TRANS + 8), F(TRANS + 9), xcur, M, DMODEL, 0);

    run_block(DEC, steQ);

    // ---- fc_out: (M,64) -> (M,1) ----
    gemm(stream, xcur, F(FCO + 0), F(FCO + 1), hb, M, DMODEL, 1);
    out_dot_kernel<<<(M + 255) / 256, 256, 0, stream>>>(hb, F(FCO + 2), F(FCO + 3),
                                                        (float*)d_out, M);
}
